// RegionTokenizer_35880156790830
// MI455X (gfx1250) — matrix operand order repaired and checked by reading, not compiled, not run on hardware
//
#include <hip/hip_runtime.h>

// ---------------------------------------------------------------------------
// RegionTokenizer for MI455X (gfx1250, wave32, WMMA bf16 16x16x32,
// async global->LDS DMA staging, double-buffered tiles)
// ---------------------------------------------------------------------------

typedef __attribute__((ext_vector_type(16))) __bf16 v16bf;
typedef __attribute__((ext_vector_type(8)))  float  v8f;

struct bf4 { __bf16 x, y, z, w; };          // 8-byte packed bf16 quad
union Frag16 { v16bf v; uint4 q[2]; };       // 16 bf16 = two 16B chunks
union CFrag  { v8f v; float f[8]; };         // 16x16 f32 accumulator (8 VGPRs)

#define B_  4
#define N_  16384
#define D_  1024
#define R_  7
#define H_  128

static __device__ inline v8f wmma_bf16(v16bf a, v16bf b, v8f c) {
  // D = A(16x32 bf16) * B(32x16 bf16) + C(16x16 f32)
  return __builtin_amdgcn_wmma_f32_16x16x32_bf16(
      /*neg_a=*/false, a, /*neg_b=*/false, b,
      /*c_mod=*/(short)0, c, /*reuse_a=*/false, /*reuse_b=*/false);
}

// Async DMA: copy 16 contiguous bytes global -> LDS, tracked by ASYNCcnt.
// VGLOBAL async encoding: VDST = LDS byte address, VADDR = 64-bit address, GV mode.
static __device__ inline void cp_async16(const __bf16* g, __bf16* l) {
  asm volatile("global_load_async_to_lds_b128 %0, %1, off"
               :: "v"((unsigned)(uintptr_t)l),
                  "v"((unsigned long long)(uintptr_t)g)
               : "memory");
}
static __device__ inline void wait_async0() {
  asm volatile("s_wait_asynccnt 0x0" ::: "memory");
}

// ---------------------------------------------------------------------------
__global__ void k_init(unsigned* p) { *p = 0u; }

// max |coords| over B*N*2 floats -> ws scalar (float bits, monotone for >=0)
__global__ __launch_bounds__(256) void k_maxabs(const float* __restrict__ crd,
                                                unsigned* __restrict__ mx) {
  __shared__ float sm[256];
  int t = threadIdx.x;
  float m = 0.f;
  for (size_t i = (size_t)blockIdx.x * 256 + t; i < (size_t)B_ * N_ * 2;
       i += (size_t)gridDim.x * 256)
    m = fmaxf(m, fabsf(crd[i]));
  sm[t] = m; __syncthreads();
  for (int s = 128; s > 0; s >>= 1) {
    if (t < s) sm[t] = fmaxf(sm[t], sm[t + s]);
    __syncthreads();
  }
  if (t == 0) atomicMax(mx, __float_as_uint(sm[0]));
}

// convert weights to bf16 AND transpose so GEMM B-tiles are contiguous in K:
//   WpT[n][k]    = W_pos[k][n]          (D x D)
//   W1T[r][h][k] = W1[r][k][h]          (R x H x D)
__global__ __launch_bounds__(256) void k_cvt(const float* __restrict__ Wp,
                                             const float* __restrict__ W1,
                                             unsigned short* WpT,
                                             unsigned short* W1T) {
  int i = blockIdx.x * 256 + threadIdx.x;
  if (i < D_ * D_) {
    int k = i >> 10, n = i & (D_ - 1);
    ((__bf16*)WpT)[(size_t)n * D_ + k] = (__bf16)Wp[i];
  }
  if (i < R_ * D_ * H_) {
    int r   = i >> 17;              // D_*H_ = 131072 = 2^17
    int rem = i & (D_ * H_ - 1);
    int k   = rem >> 7, h = rem & (H_ - 1);
    ((__bf16*)W1T)[(size_t)r * D_ * H_ + (size_t)h * D_ + k] = (__bf16)W1[i];
  }
}

// positional encoding in bf16; one block per (b,n), thread j handles pe[4j..4j+3]
__global__ __launch_bounds__(256) void k_pe(const float* __restrict__ crd,
                                            const unsigned* __restrict__ mx,
                                            unsigned short* __restrict__ peb) {
  int j  = threadIdx.x;          // 0..255 == D/4
  int bn = blockIdx.x;           // 0..B*N-1
  float ma  = __uint_as_float(*mx);
  float inv = 1.f / (ma + 1.f);
  float cx = crd[(size_t)bn * 2 + 0] * inv;
  float cy = crd[(size_t)bn * 2 + 1] * inv;
  float dt = __expf(-0.0179889470f * (float)j);   // exp(-j*ln(1e4)/512)
  float sx, cxv, sy, cyv;
  __sincosf(cx * dt, &sx, &cxv);
  __sincosf(cy * dt, &sy, &cyv);
  bf4 o; o.x = (__bf16)sx; o.y = (__bf16)cxv; o.z = (__bf16)sy; o.w = (__bf16)cyv;
  ((bf4*)peb)[(size_t)bn * (D_ / 4) + j] = o;
}

// ---------------------------------------------------------------------------
// GEMM1: x = pf + pe @ W_pos + b_pos ; M=B*N, K=D, N=D ; x stored bf16
// block tile 128x128, 8 waves (4M x 2N), wave tile 32x64 = 2x4 WMMA tiles
// A and B^T tiles staged by async DMA, double-buffered, 1 barrier / k-step
// ---------------------------------------------------------------------------
__global__ __launch_bounds__(256) void k_gemm_x(const unsigned short* __restrict__ peb_,
                                                const unsigned short* __restrict__ WpT_,
                                                const float* __restrict__ pf,
                                                const float* __restrict__ bpos,
                                                unsigned short* __restrict__ xb_) {
  const __bf16* peb = (const __bf16*)peb_;
  const __bf16* WpT = (const __bf16*)WpT_;
  __bf16* xb = (__bf16*)xb_;
  __shared__ __bf16 As[2][128 * 48];   // A tile 128x32, row stride 48 (16B align)
  __shared__ __bf16 Bs[2][128 * 48];   // B^T tile: [col(128)][k(32)], stride 48
  const int t = threadIdx.x, lane = t & 31, wave = t >> 5;
  const int wm = wave >> 1, wn = wave & 1;
  const int hf = lane >> 4, l16 = lane & 15;
  const int n0 = blockIdx.x * 128, m0 = blockIdx.y * 128;
  const int rw = t >> 2, sg = t & 3;   // per-thread 16B staging unit
  CFrag acc[2][4] = {};

  // stage k-step tiles into buffer `b` (4 async B128 per thread)
  auto stage = [&](int b, int k0) {
    cp_async16(peb + (size_t)(m0 + rw)      * D_ + k0 + sg * 8, As[b] + rw       * 48 + sg * 8);
    cp_async16(peb + (size_t)(m0 + rw + 64) * D_ + k0 + sg * 8, As[b] + (rw + 64) * 48 + sg * 8);
    cp_async16(WpT + (size_t)(n0 + rw)      * D_ + k0 + sg * 8, Bs[b] + rw       * 48 + sg * 8);
    cp_async16(WpT + (size_t)(n0 + rw + 64) * D_ + k0 + sg * 8, Bs[b] + (rw + 64) * 48 + sg * 8);
  };

  stage(0, 0);
  int buf = 0;
  for (int k0 = 0; k0 < D_; k0 += 32) {
    wait_async0();        // own DMA done
    __syncthreads();      // everyone's DMA done + prev buffer reads done
    if (k0 + 32 < D_) stage(buf ^ 1, k0 + 32);   // prefetch next k-step

    Frag16 af[2], bfv[4];
    for (int i = 0; i < 2; i++) {
      int m = wm * 32 + i * 16 + l16;
      af[i].q[0] = *(const uint4*)(As[buf] + m * 48 + hf * 16);      // K=hf*16..
      af[i].q[1] = *(const uint4*)(As[buf] + m * 48 + hf * 16 + 8);  // K=hf*16+8..
    }
    for (int j = 0; j < 4; j++) {
      int c = wn * 64 + j * 16 + l16;
      bfv[j].q[0] = *(const uint4*)(Bs[buf] + c * 48 + hf * 16);
      bfv[j].q[1] = *(const uint4*)(Bs[buf] + c * 48 + hf * 16 + 8);
    }
    for (int i = 0; i < 2; i++)
      for (int j = 0; j < 4; j++)
        acc[i][j].v = wmma_bf16(af[i].v, bfv[j].v, acc[i][j].v);
    buf ^= 1;
  }

  // epilogue: + patch_features + b_pos, store bf16 x
  for (int j = 0; j < 4; j++) {
    int col = n0 + wn * 64 + j * 16 + l16;
    float bp = bpos[col];
    for (int i = 0; i < 2; i++) {
      int rbase = m0 + wm * 32 + i * 16 + hf * 8;
      for (int v = 0; v < 8; v++) {
        size_t row = (size_t)(rbase + v);
        float val = acc[i][j].f[v] + pf[row * D_ + col] + bp;
        xb[row * D_ + col] = (__bf16)val;
      }
    }
  }
}

// ---------------------------------------------------------------------------
// GEMM2 + fused logits: logits[b,n,r] = sum_h tanh(x@W1[r]+b1[r,h])*W2[r,h] + b2[r]
// block: 64 rows x full H=128 for one region r; 8 waves (4M x 2N), wave 16x64
// ---------------------------------------------------------------------------
__global__ __launch_bounds__(256) void k_gemm_logits(const unsigned short* __restrict__ xb_,
                                                     const unsigned short* __restrict__ W1T_,
                                                     const float* __restrict__ b1,
                                                     const float* __restrict__ W2,
                                                     const float* __restrict__ b2,
                                                     float* __restrict__ logits) {
  const __bf16* xb = (const __bf16*)xb_;
  const int r = blockIdx.y;
  const __bf16* W1Tr = (const __bf16*)W1T_ + (size_t)r * D_ * H_;  // [h][k]
  __shared__ __bf16 As[2][64 * 48];
  __shared__ __bf16 Bs[2][128 * 48];
  __shared__ float  red[2][64];
  const int t = threadIdx.x, lane = t & 31, wave = t >> 5;
  const int wm = wave >> 1, wn = wave & 1;
  const int hf = lane >> 4, l16 = lane & 15;
  const int m0 = blockIdx.x * 64;
  const int rw = t >> 2, sg = t & 3;
  CFrag acc[4] = {};

  auto stage = [&](int b, int k0) {
    cp_async16(xb   + (size_t)(m0 + rw) * D_ + k0 + sg * 8, As[b] + rw       * 48 + sg * 8);
    cp_async16(W1Tr + (size_t)rw        * D_ + k0 + sg * 8, Bs[b] + rw       * 48 + sg * 8);
    cp_async16(W1Tr + (size_t)(rw + 64) * D_ + k0 + sg * 8, Bs[b] + (rw + 64) * 48 + sg * 8);
  };

  stage(0, 0);
  int buf = 0;
  for (int k0 = 0; k0 < D_; k0 += 32) {
    wait_async0();
    __syncthreads();
    if (k0 + 32 < D_) stage(buf ^ 1, k0 + 32);

    Frag16 af, bfv[4];
    int m = wm * 16 + l16;
    af.q[0] = *(const uint4*)(As[buf] + m * 48 + hf * 16);
    af.q[1] = *(const uint4*)(As[buf] + m * 48 + hf * 16 + 8);
    for (int j = 0; j < 4; j++) {
      int c = wn * 64 + j * 16 + l16;
      bfv[j].q[0] = *(const uint4*)(Bs[buf] + c * 48 + hf * 16);
      bfv[j].q[1] = *(const uint4*)(Bs[buf] + c * 48 + hf * 16 + 8);
    }
    for (int j = 0; j < 4; j++) acc[j].v = wmma_bf16(af.v, bfv[j].v, acc[j].v);
    buf ^= 1;
  }

  // fused epilogue: tanh(.+b1)*W2, reduce across H
  float p[8];
  for (int v = 0; v < 8; v++) p[v] = 0.f;
  for (int j = 0; j < 4; j++) {
    int h = wn * 64 + j * 16 + l16;
    float w2v = W2[r * H_ + h];
    float b1v = b1[r * H_ + h];
    for (int v = 0; v < 8; v++) p[v] += tanhf(acc[j].f[v] + b1v) * w2v;
  }
  for (int off = 1; off < 16; off <<= 1)           // butterfly over 16 cols
    for (int v = 0; v < 8; v++) p[v] += __shfl_xor(p[v], off, 32);
  if (l16 == 0)
    for (int v = 0; v < 8; v++) red[wn][wm * 16 + hf * 8 + v] = p[v];
  __syncthreads();
  if (t < 64) {
    size_t row = (size_t)(m0 + t);
    logits[row * R_ + r] = red[0][t] + red[1][t] + b2[r];
  }
}

// ---------------------------------------------------------------------------
// segment softmax stats per (b,r): masked max + sum(exp)
// ---------------------------------------------------------------------------
__global__ __launch_bounds__(256) void k_stats(const float* __restrict__ logits,
                                               const int* __restrict__ lab,
                                               float* __restrict__ stats) {
  const int blk = blockIdx.x, b = blk / R_, r = blk % R_;
  const int t = threadIdx.x;
  __shared__ float sm[256];
  const int* lb = lab + (size_t)b * N_;
  float mx = -3.0e38f;
  for (int n = t; n < N_; n += 256)
    if (lb[n] == r) mx = fmaxf(mx, logits[((size_t)b * N_ + n) * R_ + r]);
  sm[t] = mx; __syncthreads();
  for (int s = 128; s > 0; s >>= 1) {
    if (t < s) sm[t] = fmaxf(sm[t], sm[t + s]);
    __syncthreads();
  }
  mx = sm[0]; __syncthreads();
  float sum = 0.f;
  for (int n = t; n < N_; n += 256)
    if (lb[n] == r) sum += __expf(logits[((size_t)b * N_ + n) * R_ + r] - mx);
  sm[t] = sum; __syncthreads();
  for (int s = 128; s > 0; s >>= 1) {
    if (t < s) sm[t] += sm[t + s];
    __syncthreads();
  }
  if (t == 0) { stats[blk * 4 + 0] = mx; stats[blk * 4 + 1] = sm[0]; }
}

// ---------------------------------------------------------------------------
// weighted pool + region_emb + LayerNorm -> out ; nonempty flags appended
// one block per (b,r); thread t owns columns 4t..4t+3
// ---------------------------------------------------------------------------
__global__ __launch_bounds__(256) void k_pool_ln(const unsigned short* __restrict__ xb_,
                                                 const int* __restrict__ lab,
                                                 const float* __restrict__ logits,
                                                 const float* __restrict__ stats,
                                                 const float* __restrict__ remb,
                                                 const float* __restrict__ gamma,
                                                 const float* __restrict__ beta,
                                                 float* __restrict__ out) {
  const __bf16* xb = (const __bf16*)xb_;
  const int blk = blockIdx.x, b = blk / R_, r = blk % R_;
  const int t = threadIdx.x;
  const float mx = stats[blk * 4 + 0], S = stats[blk * 4 + 1];
  const bool  ne = (S > 0.f);
  const float invS = ne ? 1.f / S : 0.f;
  const int* lb = lab + (size_t)b * N_;
  float a0 = 0.f, a1 = 0.f, a2 = 0.f, a3 = 0.f;
  for (int n = 0; n < N_; n++) {
    if (lb[n] == r) {
      float w = __expf(logits[((size_t)b * N_ + n) * R_ + r] - mx) * invS;
      bf4 xv = *(const bf4*)(xb + ((size_t)b * N_ + n) * D_ + t * 4);
      a0 += w * (float)xv.x; a1 += w * (float)xv.y;
      a2 += w * (float)xv.z; a3 += w * (float)xv.w;
    }
  }
  int c0 = t * 4;
  float tok[4];
  tok[0] = ne ? a0 + remb[r * D_ + c0 + 0] : 0.f;
  tok[1] = ne ? a1 + remb[r * D_ + c0 + 1] : 0.f;
  tok[2] = ne ? a2 + remb[r * D_ + c0 + 2] : 0.f;
  tok[3] = ne ? a3 + remb[r * D_ + c0 + 3] : 0.f;
  __shared__ float s1[256], s2[256];
  s1[t] = tok[0] + tok[1] + tok[2] + tok[3];
  s2[t] = tok[0]*tok[0] + tok[1]*tok[1] + tok[2]*tok[2] + tok[3]*tok[3];
  __syncthreads();
  for (int s = 128; s > 0; s >>= 1) {
    if (t < s) { s1[t] += s1[t + s]; s2[t] += s2[t + s]; }
    __syncthreads();
  }
  float mu   = s1[0] * (1.f / 1024.f);
  float var  = s2[0] * (1.f / 1024.f) - mu * mu;
  float rstd = rsqrtf(var + 1e-5f);
  for (int j = 0; j < 4; j++) {
    int col = c0 + j;
    out[((size_t)(b * R_ + r)) * D_ + col] =
        (tok[j] - mu) * rstd * gamma[col] + beta[col];
  }
  if (t == 0) out[(size_t)B_ * R_ * D_ + b * R_ + r] = ne ? 1.f : 0.f;
}

// ---------------------------------------------------------------------------
extern "C" void kernel_launch(void* const* d_in, const int* in_sizes, int n_in,
                              void* d_out, int out_size, void* d_ws, size_t ws_size,
                              hipStream_t stream) {
  const float* pf   = (const float*)d_in[0];
  const float* crd  = (const float*)d_in[1];
  const int*   lab  = (const int*)  d_in[2];
  const float* Wpos = (const float*)d_in[3];
  const float* bpos = (const float*)d_in[4];
  const float* W1   = (const float*)d_in[5];
  const float* b1   = (const float*)d_in[6];
  const float* W2   = (const float*)d_in[7];
  const float* b2   = (const float*)d_in[8];
  const float* remb = (const float*)d_in[9];
  const float* gam  = (const float*)d_in[10];
  const float* bet  = (const float*)d_in[11];
  float* out = (float*)d_out;
  char*  ws  = (char*)d_ws;

  const size_t SZ_PE = (size_t)B_ * N_ * D_ * 2;      // 128 MB bf16
  unsigned*       mxp = (unsigned*)ws;
  unsigned short* peb = (unsigned short*)(ws + 256);
  unsigned short* xb  = (unsigned short*)(ws + 256 + SZ_PE);
  unsigned short* WpT = (unsigned short*)(ws + 256 + 2 * SZ_PE);
  unsigned short* W1T = (unsigned short*)(ws + 256 + 2 * SZ_PE + (size_t)D_ * D_ * 2);
  float* lg = (float*)(ws + 256 + 2 * SZ_PE + (size_t)D_ * D_ * 2 + (size_t)R_ * D_ * H_ * 2);
  float* st = (float*)((char*)lg + (size_t)B_ * N_ * R_ * 4);

  k_init<<<1, 1, 0, stream>>>(mxp);
  k_maxabs<<<128, 256, 0, stream>>>(crd, mxp);
  k_cvt<<<(D_ * D_) / 256, 256, 0, stream>>>(Wpos, W1, WpT, W1T);
  k_pe<<<B_ * N_, 256, 0, stream>>>(crd, mxp, peb);
  k_gemm_x<<<dim3(D_ / 128, (B_ * N_) / 128), 256, 0, stream>>>(peb, WpT, pf, bpos, xb);
  k_gemm_logits<<<dim3((B_ * N_) / 64, R_), 256, 0, stream>>>(xb, W1T, b1, W2, b2, lg);
  k_stats<<<B_ * R_, 256, 0, stream>>>(lg, lab, st);
  k_pool_ln<<<B_ * R_, 256, 0, stream>>>(xb, lab, lg, st, remb, gam, bet, out);
}
